// DynamicTimeWindow_1030792151094
// MI455X (gfx1250) — compile-verified, matrix operand order checked by reading, not run on hardware
//
#include <hip/hip_runtime.h>
#include <math.h>

// ---------------- problem constants ----------------
#define BN      8192
#define TT      22
#define OBS     256
#define ACTD    64
#define INDIM   320
#define HH      32
#define GG      96      // 3*H
#define TS      15      // CENTER+1 timesteps
#define FEATN   35
#define MAXW    15
#define WPB     4       // waves per block in GRU kernel (block = 128 threads)

// ---------------- WMMA vector types ----------------
typedef __bf16 bf16_t;
typedef bf16_t v16bf __attribute__((ext_vector_type(16)));
typedef bf16_t v8bf  __attribute__((ext_vector_type(8)));
typedef float  v8f   __attribute__((ext_vector_type(8)));

// ---------------- LDS layout for GRU kernel (bytes) ----------------
// padded row strides (in bf16 halves / f32 words) to spread LDS banks
constexpr int WIH_LD = 328;   // W_ih rows (K-pad 320->328)
constexpr int WHH_LD = 40;    // W_hh rows (32->40)
constexpr int XT_LD  = 328;   // x-tile rows
constexpr int HB_LD  = 40;    // h bf16 rows
constexpr int HF_LD  = 33;    // h f32 rows (scalar access)

constexpr int OFF_WIH = 0;
constexpr int OFF_WHH = OFF_WIH + GG * WIH_LD * 2;
constexpr int OFF_BIH = OFF_WHH + GG * WHH_LD * 2;
constexpr int OFF_BHH = OFF_BIH + GG * 4;
constexpr int OFF_W1  = OFF_BHH + GG * 4;
constexpr int OFF_B1  = OFF_W1  + 64 * FEATN * 4;
constexpr int OFF_W2  = OFF_B1  + 64 * 4;
constexpr int OFF_B2  = OFF_W2  + 32 * 64 * 4;
constexpr int OFF_W3  = OFF_B2  + 32 * 4;
constexpr int OFF_B3  = OFF_W3  + 14 * 32 * 4;
constexpr int OFF_LNG = OFF_B3  + 16 * 4;
constexpr int OFF_LNB = OFF_LNG + 36 * 4;
constexpr int OFF_X   = ((OFF_LNB + 36 * 4) + 15) & ~15;         // 16B aligned
constexpr int OFF_HB  = OFF_X  + WPB * 16 * XT_LD * 2;           // stays 16B aligned
constexpr int OFF_HF  = OFF_HB + WPB * 16 * HB_LD * 2;
constexpr int SMEM_BYTES = OFF_HF + WPB * 16 * HF_LD * 4;        // ~143 KB < 320 KB/WGP

// ---------------- wave32 reductions ----------------
__device__ __forceinline__ float wsum(float v) {
#pragma unroll
  for (int m = 16; m > 0; m >>= 1) v += __shfl_xor(v, m, 32);
  return v;
}
__device__ __forceinline__ float wmaxr(float v) {
#pragma unroll
  for (int m = 16; m > 0; m >>= 1) v = fmaxf(v, __shfl_xor(v, m, 32));
  return v;
}
__device__ __forceinline__ float sigm(float x) { return 1.f / (1.f + expf(-x)); }

// Fragment loader: row-major bf16 tile in LDS -> CDNA5 16-bit A/B operand layout.
// lane half h: v16bf[0..7] = K k0+8h..k0+8h+7, v16bf[8..15] = K k0+16+8h..k0+16+8h+7
__device__ __forceinline__ v16bf load_frag(const bf16_t* __restrict__ base,
                                           int row, int ld, int k0, int half) {
  const bf16_t* p = base + row * ld + k0 + 8 * half;
  union { v16bf v; v8bf h[2]; } u;
  u.h[0] = *(const v8bf*)(p);
  u.h[1] = *(const v8bf*)(p + 16);
  return u.v;
}

// =====================================================================
// Kernel 1: entropy / rate_of_change / correlation  (one wave per row)
// =====================================================================
__global__ __launch_bounds__(256)
void dtw_features_kernel(const float* __restrict__ obs,
                         const float* __restrict__ act,
                         float* __restrict__ feats3) {
  const int lane = threadIdx.x & 31;
  const long b = (long)blockIdx.x * 8 + (threadIdx.x >> 5);

  const float* o14 = obs + (b * TT + 14) * OBS;
  float x[8];
#pragma unroll
  for (int i = 0; i < 8; ++i) x[i] = o14[lane + 32 * i];

  // ---- softmax entropy over obs_t ----
  float mx = x[0];
#pragma unroll
  for (int i = 1; i < 8; ++i) mx = fmaxf(mx, x[i]);
  mx = wmaxr(mx);
  float e[8], se = 0.f;
#pragma unroll
  for (int i = 0; i < 8; ++i) { e[i] = expf(x[i] - mx); se += e[i]; }
  se = wsum(se);
  float ent = 0.f;
#pragma unroll
  for (int i = 0; i < 8; ++i) { float p = e[i] / se; ent += p * logf(p + 1e-8f); }
  ent = -wsum(ent);

  // ---- rate of change: mean of ||obs[t]-obs[t-1]|| for t = 12,13,14 ----
  float rate = 0.f;
#pragma unroll
  for (int d = 0; d < 3; ++d) {
    const float* ra = obs + (b * TT + 12 + d) * OBS;
    const float* rb = obs + (b * TT + 11 + d) * OBS;
    float s = 0.f;
    for (int i = lane; i < OBS; i += 32) { float df = ra[i] - rb[i]; s += df * df; }
    rate += sqrtf(wsum(s));
  }
  rate *= (1.f / 3.f);

  // ---- correlation(obs_t, pad(act[t=13])) ----
  float so = 0.f;
#pragma unroll
  for (int i = 0; i < 8; ++i) so += x[i];
  const float mo = wsum(so) * (1.f / 256.f);
  const float* ap = act + (b * TT + 13) * ACTD;
  const float ma = wsum(ap[lane] + ap[lane + 32]) * (1.f / 256.f);
  float num = 0.f, no = 0.f, na = 0.f;
#pragma unroll
  for (int i = 0; i < 8; ++i) {
    const int idx = lane + 32 * i;
    const float oc = x[i] - mo;
    const float av = (idx < ACTD ? ap[idx] : 0.f) - ma;
    num += oc * av; no += oc * oc; na += av * av;
  }
  num = wsum(num); no = wsum(no); na = wsum(na);
  const float corr = num / (sqrtf(no) * sqrtf(na) + 1e-8f);

  if (lane == 0) {
    feats3[b * 3 + 0] = ent;
    feats3[b * 3 + 1] = rate;
    feats3[b * 3 + 2] = corr;
  }
}

// =====================================================================
// Kernel 2: WMMA bf16 GRU (15 steps) + LayerNorm + MLP + argmax
// one wave = 16 batch rows; block = WPB waves
// =====================================================================
__global__ __launch_bounds__(WPB * 32)
void dtw_gru_mlp_kernel(const float* __restrict__ obs, const float* __restrict__ act,
                        const float* __restrict__ Wih, const float* __restrict__ Whh,
                        const float* __restrict__ bih, const float* __restrict__ bhh,
                        const float* __restrict__ lng, const float* __restrict__ lnb,
                        const float* __restrict__ W1, const float* __restrict__ b1,
                        const float* __restrict__ W2, const float* __restrict__ b2,
                        const float* __restrict__ W3, const float* __restrict__ b3,
                        const float* __restrict__ feats3,
                        int* __restrict__ wlenI, float* __restrict__ wlenF) {
  extern __shared__ __align__(16) char smem[];
  bf16_t* sWih = (bf16_t*)(smem + OFF_WIH);
  bf16_t* sWhh = (bf16_t*)(smem + OFF_WHH);
  float*  sBih = (float*)(smem + OFF_BIH);
  float*  sBhh = (float*)(smem + OFF_BHH);
  float*  sW1  = (float*)(smem + OFF_W1);
  float*  sB1  = (float*)(smem + OFF_B1);
  float*  sW2  = (float*)(smem + OFF_W2);
  float*  sB2  = (float*)(smem + OFF_B2);
  float*  sW3  = (float*)(smem + OFF_W3);
  float*  sB3  = (float*)(smem + OFF_B3);
  float*  sLng = (float*)(smem + OFF_LNG);
  float*  sLnb = (float*)(smem + OFF_LNB);
  bf16_t* sX   = (bf16_t*)(smem + OFF_X);
  bf16_t* sHb  = (bf16_t*)(smem + OFF_HB);
  float*  sHf  = (float*)(smem + OFF_HF);

  const int tid = threadIdx.x;
  // ---- stage weights (bf16 for WMMA B-operands, f32 for MLP/LN) ----
  for (int i = tid; i < GG * INDIM; i += blockDim.x) {
    int g = i / INDIM, k = i - g * INDIM;
    sWih[g * WIH_LD + k] = (bf16_t)Wih[i];
  }
  for (int i = tid; i < GG * HH; i += blockDim.x) {
    int g = i / HH, k = i - g * HH;
    sWhh[g * WHH_LD + k] = (bf16_t)Whh[i];
  }
  for (int i = tid; i < GG; i += blockDim.x) { sBih[i] = bih[i]; sBhh[i] = bhh[i]; }
  for (int i = tid; i < 64 * FEATN; i += blockDim.x) sW1[i] = W1[i];
  for (int i = tid; i < 64; i += blockDim.x) sB1[i] = b1[i];
  for (int i = tid; i < 32 * 64; i += blockDim.x) sW2[i] = W2[i];
  for (int i = tid; i < 32; i += blockDim.x) sB2[i] = b2[i];
  for (int i = tid; i < 14 * 32; i += blockDim.x) sW3[i] = W3[i];
  for (int i = tid; i < 14; i += blockDim.x) sB3[i] = b3[i];
  for (int i = tid; i < FEATN; i += blockDim.x) { sLng[i] = lng[i]; sLnb[i] = lnb[i]; }
  for (int i = tid; i < WPB * 16 * HB_LD; i += blockDim.x) sHb[i] = (bf16_t)0.f;
  __syncthreads();

  const int w = tid >> 5, lane = tid & 31;
  const int ln = lane & 15, half = lane >> 4;
  const long rowbase = ((long)blockIdx.x * WPB + w) * 16;
  bf16_t* xb = sX  + w * 16 * XT_LD;
  bf16_t* hb = sHb + w * 16 * HB_LD;

  // ---- biases folded into the gate combine (not into accumulators) ----
  const float bR0 = sBih[ln]      + sBhh[ln];
  const float bR1 = sBih[16 + ln] + sBhh[16 + ln];
  const float bZ0 = sBih[32 + ln] + sBhh[32 + ln];
  const float bZ1 = sBih[48 + ln] + sBhh[48 + ln];
  const float bI0 = sBih[64 + ln], bH0 = sBhh[64 + ln];
  const float bI1 = sBih[80 + ln], bH1 = sBhh[80 + ln];

  // ---- W_hh B-fragments are loop-invariant: hoist all 6 ----
  v16bf whhB[6];
#pragma unroll
  for (int n = 0; n < 6; ++n)
    whhB[n] = load_frag(sWhh, n * 16 + ln, WHH_LD, 0, half);

  const v8f vzero = {0.f, 0.f, 0.f, 0.f, 0.f, 0.f, 0.f, 0.f};
  v8f hold0 = vzero, hold1 = vzero;

  for (int t = 0; t < TS; ++t) {
    // ---- stage x_t tile (16 x 320) as bf16: float4-pair loads, b128 LDS stores ----
    // unit = 8 consecutive floats; 40 units/row * 16 rows = 640 units; 20 per lane
    for (int u = lane; u < 16 * 40; u += 32) {
      const int r = u / 40, q = u - r * 40;
      const long b = rowbase + r;
      const float4* src = (q < 32)
          ? (const float4*)(obs + (b * TT + t) * OBS) + 2 * q
          : (const float4*)(act + (b * TT + t) * ACTD) + 2 * (q - 32);
      const float4 f0 = src[0], f1 = src[1];
      v8bf pk;
      pk[0] = (bf16_t)f0.x; pk[1] = (bf16_t)f0.y;
      pk[2] = (bf16_t)f0.z; pk[3] = (bf16_t)f0.w;
      pk[4] = (bf16_t)f1.x; pk[5] = (bf16_t)f1.y;
      pk[6] = (bf16_t)f1.z; pk[7] = (bf16_t)f1.w;
      *(v8bf*)(xb + r * XT_LD + 8 * q) = pk;
    }
    __syncthreads();

    v8f gi[6], gh[6];
    // ---- k0 = 0 peeled: C = inline zero (v_wmma ..., 0), no splat movs ----
    {
      const v16bf a0 = load_frag(xb, ln, XT_LD, 0, half);
#pragma unroll
      for (int n = 0; n < 6; ++n) {
        const v16bf bm = load_frag(sWih, n * 16 + ln, WIH_LD, 0, half);
        gi[n] = __builtin_amdgcn_wmma_f32_16x16x32_bf16(
            false, a0, false, bm, (short)0, vzero, false, false);
      }
      // gh = h @ W_hh^T : K=32, one k-tile, hoisted B-fragments
      const v16bf ah = load_frag(hb, ln, HB_LD, 0, half);
#pragma unroll
      for (int n = 0; n < 6; ++n)
        gh[n] = __builtin_amdgcn_wmma_f32_16x16x32_bf16(
            false, ah, false, whhB[n], (short)0, vzero, false, false);
    }
    // ---- gi accumulate over remaining 9 k-tiles ----
#pragma unroll
    for (int k0 = 1; k0 < 10; ++k0) {
      const v16bf a = load_frag(xb, ln, XT_LD, k0 * 32, half);
#pragma unroll
      for (int n = 0; n < 6; ++n) {
        const v16bf bm = load_frag(sWih, n * 16 + ln, WIH_LD, k0 * 32, half);
        gi[n] = __builtin_amdgcn_wmma_f32_16x16x32_bf16(
            false, a, false, bm, (short)0, gi[n], false, false);
      }
    }

    // ---- GRU gate combine in C-fragment layout; update h (regs + LDS bf16) ----
#pragma unroll
    for (int r = 0; r < 8; ++r) {
      const int m = r + 8 * half;
      {
        const float rg = sigm(gi[0][r] + gh[0][r] + bR0);
        const float zg = sigm(gi[2][r] + gh[2][r] + bZ0);
        const float ng = tanhf(gi[4][r] + bI0 + rg * (gh[4][r] + bH0));
        const float hn = (1.f - zg) * ng + zg * hold0[r];
        hold0[r] = hn;
        hb[m * HB_LD + ln] = (bf16_t)hn;
      }
      {
        const float rg = sigm(gi[1][r] + gh[1][r] + bR1);
        const float zg = sigm(gi[3][r] + gh[3][r] + bZ1);
        const float ng = tanhf(gi[5][r] + bI1 + rg * (gh[5][r] + bH1));
        const float hn = (1.f - zg) * ng + zg * hold1[r];
        hold1[r] = hn;
        hb[m * HB_LD + 16 + ln] = (bf16_t)hn;
      }
    }
    __syncthreads();
  }

  // ---- spill final h (f32) so each row's lane can gather it ----
  float* hf = sHf + w * 16 * HF_LD;
#pragma unroll
  for (int r = 0; r < 8; ++r) {
    const int m = r + 8 * half;
    hf[m * HF_LD + ln]      = hold0[r];
    hf[m * HF_LD + 16 + ln] = hold1[r];
  }
  __syncthreads();

  // ---- per-row tail: LayerNorm + MLP + argmax (lanes 0..15, one row each) ----
  if (lane < 16) {
    const long b = rowbase + lane;
    float f[FEATN];
    f[0] = feats3[b * 3 + 0];
    f[1] = feats3[b * 3 + 1];
    f[2] = feats3[b * 3 + 2];
#pragma unroll
    for (int j = 0; j < HH; ++j) f[3 + j] = hf[lane * HF_LD + j];

    float mu = 0.f;
#pragma unroll
    for (int k = 0; k < FEATN; ++k) mu += f[k];
    mu *= (1.f / (float)FEATN);
    float var = 0.f;
#pragma unroll
    for (int k = 0; k < FEATN; ++k) { const float d = f[k] - mu; var += d * d; }
    var *= (1.f / (float)FEATN);
    const float inv = rsqrtf(var + 1e-5f);
#pragma unroll
    for (int k = 0; k < FEATN; ++k) f[k] = (f[k] - mu) * inv * sLng[k] + sLnb[k];

    float h1[64];
    for (int o = 0; o < 64; ++o) {
      float s = sB1[o];
      const float* wr = sW1 + o * FEATN;
#pragma unroll
      for (int k = 0; k < FEATN; ++k) s += f[k] * wr[k];
      h1[o] = fmaxf(s, 0.f);
    }
    float h2[32];
    for (int o = 0; o < 32; ++o) {
      float s = sB2[o];
      const float* wr = sW2 + o * 64;
#pragma unroll
      for (int k = 0; k < 64; ++k) s += h1[k] * wr[k];
      h2[o] = fmaxf(s, 0.f);
    }
    float best = -INFINITY; int bi = 0;
    for (int o = 0; o < 14; ++o) {
      float s = sB3[o];
      const float* wr = sW3 + o * 32;
#pragma unroll
      for (int k = 0; k < 32; ++k) s += h2[k] * wr[k];
      if (s > best) { best = s; bi = o; }
    }
    const int wl = bi + 2;
    wlenI[b] = wl;
    wlenF[b] = (float)wl;
  }
}

// =====================================================================
// Kernel 3: streaming masked window write (float4) + mask output
// =====================================================================
__global__ __launch_bounds__(256)
void dtw_window_kernel(const float* __restrict__ obs, const float* __restrict__ act,
                       const int* __restrict__ wlen,
                       float* __restrict__ out_pw, float* __restrict__ out_mask) {
  const long NU = (long)BN * MAXW * 80;   // 80 float4 per (b, j)
  const long stride = (long)gridDim.x * blockDim.x;
  for (long u = (long)blockIdx.x * blockDim.x + threadIdx.x; u < NU; u += stride) {
    const long b = u / (MAXW * 80);
    const int rem = (int)(u - b * (MAXW * 80));
    const int j = rem / 80, cq = rem - j * 80;

    const int wl = wlen[b];
    const int soff = (wl - 1) >> 1, eoff = wl >> 1;
    const int off = j - 7;
    const float m = (off >= -soff && off <= eoff) ? 1.f : 0.f;

    const float* src = (cq < 64)
        ? obs + (b * TT + 7 + j) * OBS + 4 * cq
        : act + (b * TT + 7 + j) * ACTD + (4 * cq - OBS);
    float4 v = *(const float4*)src;
    v.x *= m; v.y *= m; v.z *= m; v.w *= m;
    *(float4*)(out_pw + (b * MAXW + j) * INDIM + 4 * cq) = v;
    if (cq == 0) out_mask[b * MAXW + j] = m;
  }
}

// =====================================================================
extern "C" void kernel_launch(void* const* d_in, const int* in_sizes, int n_in,
                              void* d_out, int out_size, void* d_ws, size_t ws_size,
                              hipStream_t stream) {
  (void)in_sizes; (void)n_in; (void)out_size; (void)ws_size;
  const float* obs = (const float*)d_in[0];
  const float* act = (const float*)d_in[1];
  const float* Wih = (const float*)d_in[2];
  const float* Whh = (const float*)d_in[3];
  const float* bih = (const float*)d_in[4];
  const float* bhh = (const float*)d_in[5];
  const float* lng = (const float*)d_in[6];
  const float* lnb = (const float*)d_in[7];
  const float* W1  = (const float*)d_in[8];
  const float* b1  = (const float*)d_in[9];
  const float* W2  = (const float*)d_in[10];
  const float* b2  = (const float*)d_in[11];
  const float* W3  = (const float*)d_in[12];
  const float* b3  = (const float*)d_in[13];

  float* out_wlen = (float*)d_out;                       // (B,) window_len as float
  float* out_pw   = out_wlen + BN;                       // (B,15,320)
  float* out_mask = out_pw + (long)BN * MAXW * INDIM;    // (B,15)

  float* feats3 = (float*)d_ws;                          // (B,3)
  int*   wlenI  = (int*)((char*)d_ws + (size_t)BN * 3 * sizeof(float));

  dtw_features_kernel<<<BN / 8, 256, 0, stream>>>(obs, act, feats3);

  dtw_gru_mlp_kernel<<<BN / (16 * WPB), WPB * 32, SMEM_BYTES, stream>>>(
      obs, act, Wih, Whh, bih, bhh, lng, lnb,
      W1, b1, W2, b2, W3, b3, feats3, wlenI, out_wlen);

  dtw_window_kernel<<<2048, 256, 0, stream>>>(obs, act, wlenI, out_pw, out_mask);
}
